// st_gcn_57784490000385
// MI455X (gfx1250) — matrix operand order repaired and checked by reading, not compiled, not run on hardware
//
#include <hip/hip_runtime.h>
#include <hip/hip_bf16.h>

// Sizes from the reference
#define N_    16
#define CIN_  16
#define T_    512
#define V_    200
#define K_    3
#define COUT_ 64
#define H_    64

typedef __attribute__((ext_vector_type(16))) _Float16 v16h;
typedef __attribute__((ext_vector_type(8)))  float    v8f;

// ---------------------------------------------------------------------------
// P1: s[k][v] = (1/V) * sum_w A[k,v,w]      (600 outputs)
// ---------------------------------------------------------------------------
__global__ void k_s(const float* __restrict__ A, float* __restrict__ s) {
  int id = blockIdx.x * 256 + threadIdx.x;
  if (id >= K_ * V_) return;
  int k = id / V_, v = id % V_;
  const float* row = A + ((long)k * V_ + v) * V_;
  float acc = 0.f;
  for (int w = 0; w < V_; ++w) acc += row[w];
  s[id] = acc * (1.0f / V_);
}

// ---------------------------------------------------------------------------
// P2: beff[c] = sum_k b_gcn[k*64+c] * (sum_v s[k][v])
// ---------------------------------------------------------------------------
__global__ void k_beff(const float* __restrict__ s, const float* __restrict__ b_gcn,
                       float* __restrict__ beff) {
  __shared__ float asum[K_];
  int tid = threadIdx.x;
  if (tid < K_) {
    float a = 0.f;
    for (int v = 0; v < V_; ++v) a += s[tid * V_ + v];
    asum[tid] = a;
  }
  __syncthreads();
  if (tid < COUT_) {
    float b = 0.f;
    for (int k = 0; k < K_; ++k) b += b_gcn[k * COUT_ + tid] * asum[k];
    beff[tid] = b;
  }
}

// ---------------------------------------------------------------------------
// P3: Wz[(cin*3+k)*64 + c] = W_gcn[(k*64+c)*16 + cin]     (48x64 repack)
// ---------------------------------------------------------------------------
__global__ void k_wz(const float* __restrict__ W_gcn, float* __restrict__ Wz) {
  int id = blockIdx.x * 256 + threadIdx.x;
  if (id >= 48 * COUT_) return;
  int r = id >> 6, c = id & 63;
  int cin = r / 3, k = r % 3;
  Wz[id] = W_gcn[(k * COUT_ + c) * CIN_ + cin];
}

// ---------------------------------------------------------------------------
// B1: u[t][n][cin*3+k] = sum_v x[n,cin,t,v]*s[k,v]
// One wave32 per (n,cin,t) row (streams x once: 104.8 MB -> HBM-bound).
// ---------------------------------------------------------------------------
__global__ void __launch_bounds__(256) k_u(const float* __restrict__ x,
                                           const float* __restrict__ s_g,
                                           float* __restrict__ U) {
  __shared__ float s[K_ * V_];
  int tid = threadIdx.x;
  for (int i = tid; i < K_ * V_; i += 256) s[i] = s_g[i];
  __syncthreads();
  int wave = tid >> 5, lane = tid & 31;
  long rid = (long)blockIdx.x * 8 + wave;          // 16*16*512 = 131072 rows
  const float* xr = x + rid * V_;
  float p0 = 0.f, p1 = 0.f, p2 = 0.f;
  for (int v = lane; v < V_; v += 32) {
    float xv = xr[v];
    p0 += xv * s[0 * V_ + v];
    p1 += xv * s[1 * V_ + v];
    p2 += xv * s[2 * V_ + v];
  }
  for (int off = 16; off > 0; off >>= 1) {         // wave32 reduction
    p0 += __shfl_xor(p0, off, 32);
    p1 += __shfl_xor(p1, off, 32);
    p2 += __shfl_xor(p2, off, 32);
  }
  if (lane == 0) {
    int n   = (int)(rid / (CIN_ * T_));
    int rem = (int)(rid % (CIN_ * T_));
    int cin = rem / T_;
    int t   = rem % T_;
    float* u = U + ((long)(t * N_ + n)) * 48 + cin * 3;
    u[0] = p0; u[1] = p1; u[2] = p2;
  }
}

// ---------------------------------------------------------------------------
// B2: z[t][n][c] = sum_j U[t][n][j]*Wz[j][c] + beff[c]; store as f16 for WMMA
// ---------------------------------------------------------------------------
__global__ void __launch_bounds__(256) k_z(const float* __restrict__ U,
                                           const float* __restrict__ Wz,
                                           const float* __restrict__ beff,
                                           _Float16* __restrict__ zh) {
  __shared__ float wz[48 * COUT_];
  __shared__ float be[COUT_];
  __shared__ float ur[4 * 48];
  int tid = threadIdx.x;
  for (int i = tid; i < 48 * COUT_; i += 256) wz[i] = Wz[i];
  if (tid < COUT_) be[tid] = beff[tid];
  long base_row = (long)blockIdx.x * 4;            // rows = t*16+n, 8192 total
  for (int i = tid; i < 4 * 48; i += 256) ur[i] = U[base_row * 48 + i];
  __syncthreads();
  int rl = tid >> 6, c = tid & 63;
  float acc = be[c];
#pragma unroll
  for (int j = 0; j < 48; ++j) acc += ur[rl * 48 + j] * wz[j * COUT_ + c];
  zh[(base_row + rl) * COUT_ + c] = (_Float16)acc;
}

// ---------------------------------------------------------------------------
// WMMA fragment builders (ISA 7.12.2 layouts; compile to 2x b128 per frag)
// ---------------------------------------------------------------------------
__device__ __forceinline__ v16h frag_a(const _Float16* base, int lane, int koff) {
  int m  = lane & 15;
  int kg = (lane >> 4) * 8;
  v16h a;
#pragma unroll
  for (int j = 0; j < 8; ++j) {
    int kb = ((j < 4) ? 0 : 16) + kg + 2 * (j & 3) + koff;
    a[2 * j]     = base[m * 64 + kb];
    a[2 * j + 1] = base[m * 64 + kb + 1];
  }
  return a;
}

__device__ __forceinline__ v16h frag_b(const float* W, int o, int lane, int koff) {
  // W is (256 x 64) row-major; B[k][o] = W[o*64+k]
  int kg = (lane >> 4) * 8;
  v16h b;
#pragma unroll
  for (int j = 0; j < 8; ++j) {
    int kb = ((j < 4) ? 0 : 16) + kg + 2 * (j & 3) + koff;
    b[2 * j]     = (_Float16)W[o * 64 + kb];
    b[2 * j + 1] = (_Float16)W[o * 64 + kb + 1];
  }
  return b;
}

// ---------------------------------------------------------------------------
// G: time-parallel gate half: gz = z_t @ W_ih^T + b_ih + b_hh for ALL t.
// Stored in C-FRAGMENT layout (8 contiguous floats per lane) so the
// sequential kernel re-loads it as two b128s — layout is private between
// k_gz and k_lstm, which use the identical wave/lane mapping.
// ---------------------------------------------------------------------------
__global__ void __launch_bounds__(512) k_gz(const _Float16* __restrict__ zh,
                                            const float* __restrict__ Wih,
                                            const float* __restrict__ bih,
                                            const float* __restrict__ bhh,
                                            float* __restrict__ gz) {
  int tid = threadIdx.x;
  int wave = tid >> 5, lane = tid & 31;
  int t = blockIdx.x;
  int o_global = wave * 16 + (lane & 15);
  v16h b0  = frag_b(Wih, o_global, lane, 0);
  v16h b1  = frag_b(Wih, o_global, lane, 32);
  v16h az0 = frag_a(zh + (long)t * 1024, lane, 0);
  v16h az1 = frag_a(zh + (long)t * 1024, lane, 32);
  v8f acc = {};
  acc = __builtin_amdgcn_wmma_f32_16x16x32_f16(false, az0, false, b0, (short)0, acc, false, false);
  acc = __builtin_amdgcn_wmma_f32_16x16x32_f16(false, az1, false, b1, (short)0, acc, false, false);
  float bias = bih[o_global] + bhh[o_global];
  float* g = gz + (((long)t * 16 + wave) * 32 + lane) * 8;  // fragment layout
#pragma unroll
  for (int r = 0; r < 8; ++r) g[r] = acc[r] + bias;
}

// ---------------------------------------------------------------------------
// Fast activations: v_exp_f32 + v_rcp_f32 (no IEEE divide / tanhf libcall on
// the sequential critical path)
// ---------------------------------------------------------------------------
__device__ __forceinline__ float fast_sigmoid(float x) {
  return __builtin_amdgcn_rcpf(1.f + __expf(-x));
}
__device__ __forceinline__ float fast_tanh(float x) {
  return 1.f - 2.f * __builtin_amdgcn_rcpf(__expf(2.f * x) + 1.f);
}

// ---------------------------------------------------------------------------
// Sequential LSTM recurrence: per step only h @ W_hh^T (2 WMMAs per wave).
// gz accumulator is register double-buffered: the load for step t+1 issues
// during step t, hiding its L2 latency behind the WMMAs/barriers.
// ---------------------------------------------------------------------------
__global__ void __launch_bounds__(512) k_lstm(
    const float* __restrict__ Whh, const float* __restrict__ gz,
    const float* __restrict__ Wfc, const float* __restrict__ bfc,
    float* __restrict__ out) {
  __shared__ _Float16 hbuf[16 * 64];        // h state (f16 for WMMA A)
  __shared__ float    cbuf[16 * 64];        // cell state
  __shared__ float    gbuf[4 * 16 * 64];    // pre-activation gates
  __shared__ float    hflt[16 * 64];        // final-step h (f32 for FC)
  int tid  = threadIdx.x;
  int wave = tid >> 5, lane = tid & 31;
  int o_global = wave * 16 + (lane & 15);   // gate column 0..255
  int gate = wave >> 2;                     // 0=i 1=f 2=g 3=o
  int hu   = (wave & 3) * 16 + (lane & 15); // hidden unit 0..63
  int mrow = (lane >> 4) * 8;               // C/D row base for this lane half

  // Recurrent-weight B fragments live in VGPRs across all 512 steps
  v16h bhh0 = frag_b(Whh, o_global, lane, 0);
  v16h bhh1 = frag_b(Whh, o_global, lane, 32);

  const float* gbase = gz + ((long)wave * 32 + lane) * 8;

  for (int e = tid; e < 1024; e += 512) {
    hbuf[e] = (_Float16)0.f;
    cbuf[e] = 0.f;
  }

  // Preload gz fragment for t=0 (two b128 loads)
  v8f accn;
#pragma unroll
  for (int r = 0; r < 8; ++r) accn[r] = gbase[r];

  __syncthreads();

  for (int t = 0; t < T_; ++t) {
    v8f acc = accn;
    if (t + 1 < T_) {                       // issue next step's load early
      const float* gn = gbase + (long)(t + 1) * 4096;
#pragma unroll
      for (int r = 0; r < 8; ++r) accn[r] = gn[r];
    }
    v16h ah0 = frag_a(hbuf, lane, 0);
    v16h ah1 = frag_a(hbuf, lane, 32);
    acc = __builtin_amdgcn_wmma_f32_16x16x32_f16(false, ah0, false, bhh0, (short)0, acc, false, false);
    acc = __builtin_amdgcn_wmma_f32_16x16x32_f16(false, ah1, false, bhh1, (short)0, acc, false, false);
#pragma unroll
    for (int r = 0; r < 8; ++r)
      gbuf[gate * 1024 + (r + mrow) * 64 + hu] = acc[r];
    __syncthreads();
    for (int e = tid; e < 1024; e += 512) {
      float gi = fast_sigmoid(gbuf[0 * 1024 + e]);
      float gf = fast_sigmoid(gbuf[1 * 1024 + e]);
      float gg = fast_tanh   (gbuf[2 * 1024 + e]);
      float go = fast_sigmoid(gbuf[3 * 1024 + e]);
      float cc = gf * cbuf[e] + gi * gg;
      cbuf[e] = cc;
      float hv = go * fast_tanh(cc);
      hbuf[e] = (_Float16)hv;
      if (t == T_ - 1) hflt[e] = hv;
    }
    __syncthreads();
  }

  if (tid < 16) {                           // out[n] = h_last @ W_fc^T + b_fc
    float acc = bfc[0];
    for (int j = 0; j < 64; ++j) acc += hflt[tid * 64 + j] * Wfc[j];
    out[tid] = acc;
  }
}

// ---------------------------------------------------------------------------
extern "C" void kernel_launch(void* const* d_in, const int* in_sizes, int n_in,
                              void* d_out, int out_size, void* d_ws, size_t ws_size,
                              hipStream_t stream) {
  const float* x     = (const float*)d_in[0];
  const float* A     = (const float*)d_in[1];
  const float* W_gcn = (const float*)d_in[2];
  const float* b_gcn = (const float*)d_in[3];
  const float* W_ih  = (const float*)d_in[4];
  const float* W_hh  = (const float*)d_in[5];
  const float* b_ih  = (const float*)d_in[6];
  const float* b_hh  = (const float*)d_in[7];
  const float* W_fc  = (const float*)d_in[8];
  const float* b_fc  = (const float*)d_in[9];
  float* out = (float*)d_out;

  float* ws   = (float*)d_ws;
  float* s    = ws + 0;       // 600
  float* beff = ws + 640;     // 64
  float* Wz   = ws + 704;     // 3072
  float* U    = ws + 4096;    // 512*16*48 = 393216
  _Float16* zh = (_Float16*)((char*)d_ws + (size_t)(4096 + 393216) * sizeof(float)); // 1 MB
  float* gz   = (float*)((char*)d_ws + (size_t)(4096 + 393216) * sizeof(float)
                                     + (size_t)T_ * N_ * COUT_ * sizeof(_Float16)); // 8 MB

  k_s   <<<3,     256, 0, stream>>>(A, s);
  k_beff<<<1,      64, 0, stream>>>(s, b_gcn, beff);
  k_wz  <<<12,    256, 0, stream>>>(W_gcn, Wz);
  k_u   <<<16384, 256, 0, stream>>>(x, s, U);
  k_z   <<<2048,  256, 0, stream>>>(U, Wz, beff, zh);
  k_gz  <<<512,   512, 0, stream>>>(zh, W_ih, b_ih, b_hh, gz);
  k_lstm<<<1,     512, 0, stream>>>(W_hh, gz, W_fc, b_fc, out);

  // Second tuple output: passthrough of A
  hipMemcpyAsync(out + 16, A, (size_t)(K_ * V_ * V_) * sizeof(float),
                 hipMemcpyDeviceToDevice, stream);
}
